// hetero_effect_graph_27925877358636
// MI455X (gfx1250) — compile-verified, matrix operand order checked by reading, not run on hardware
//
#include <hip/hip_runtime.h>
#include <hip/hip_bf16.h>

#define NE 10000
#define NM 2048
#define D  128
#define NR 5            // relations 1..5 carry edges
#define KST (NM / 32)   // 64 K-steps over moles
#define CT 16           // 16 column tiles of concatenated X = [x_m | m1]  (256 cols)

typedef __attribute__((ext_vector_type(16))) _Float16 v16h;
typedef __attribute__((ext_vector_type(8)))  _Float16 v8h;
typedef __attribute__((ext_vector_type(8)))  float    v8f;

// K-index map for 16-bit A/B fragments (wave32):
// half h of lane L holds K = h + (h>=8 ? 8 : 0) + ((L>=16) ? 8 : 0)
__device__ __forceinline__ int kmap(int h, int hi) {
    return h + ((h >= 8) ? 8 : 0) + hi * 8;
}

// ---------------------------------------------------------------------------
// Kernel 0a: pre-swizzle weight matrices into f16 B-fragments.
// mats: 0..4 = W1[1..5], 5..9 = W2[1..5], 10 = root1, 11 = root2
// Layout: wfrag[mat][kk(4)][ct(8)][lane][16 halves]
// ---------------------------------------------------------------------------
__global__ void k_wfrag(const float* __restrict__ W1, const float* __restrict__ W2,
                        const float* __restrict__ root1, const float* __restrict__ root2,
                        _Float16* __restrict__ wfrag) {
    int lane = threadIdx.x;
    int b    = blockIdx.x;                // mat*32 + kk*8 + ct
    int ct   = b & 7;
    int kk   = (b >> 3) & 3;
    int mat  = b >> 5;
    const float* src;
    if      (mat < 5)   src = W1 + (size_t)(mat + 1) * D * D;
    else if (mat < 10)  src = W2 + (size_t)(mat - 4) * D * D;
    else if (mat == 10) src = root1;
    else                src = root2;
    int row = lane & 15;
    int hi  = lane >> 4;
    int n   = ct * 16 + row;
    v16h v;
    #pragma unroll
    for (int h = 0; h < 16; ++h) {
        int k = kk * 32 + kmap(h, hi);
        v[h] = (_Float16)src[(size_t)k * D + n];
    }
    *(v16h*)(wfrag + ((size_t)b * 32 + lane) * 16) = v;
}

// build a 16x32 f16 A-fragment from a row-major f32 matrix row
__device__ __forceinline__ v16h load_a_frag_f32(const float* rowp) {
    float4 a0 = *(const float4*)(rowp);
    float4 a1 = *(const float4*)(rowp + 4);
    float4 c0 = *(const float4*)(rowp + 16);
    float4 c1 = *(const float4*)(rowp + 20);
    float ev[16] = {a0.x, a0.y, a0.z, a0.w, a1.x, a1.y, a1.z, a1.w,
                    c0.x, c0.y, c0.z, c0.w, c1.x, c1.y, c1.z, c1.w};
    v16h ae;
    #pragma unroll
    for (int h = 0; h < 16; ++h) ae[h] = (_Float16)ev[h];
    return ae;
}

// ---------------------------------------------------------------------------
// Kernel 0b: m1 = relu(x_m @ root1 + b1) via WMMA (reuses root1 fragments).
// 128 threads = 4 waves; wave w -> output column tiles {w, w+4}.
// ---------------------------------------------------------------------------
__global__ void __launch_bounds__(128)
k_m1(const float* __restrict__ xm, const _Float16* __restrict__ wfrag,
     const float* __restrict__ b1, float* __restrict__ m1f) {
    const int lane = threadIdx.x & 31;
    const int wv   = threadIdx.x >> 5;
    const int row  = lane & 15;
    const int hi   = lane >> 4;
    const int mt   = blockIdx.x;                   // 16-mole tile

    const float* xr = xm + (size_t)(mt * 16 + row) * D + hi * 8;
    v8f a0 = (v8f)(0.0f), a1 = (v8f)(0.0f);
    #pragma unroll
    for (int kk = 0; kk < 4; ++kk) {
        v16h ae = load_a_frag_f32(xr + kk * 32);
        const _Float16* rf = wfrag + (((size_t)10 * 4 + kk) * 8) * 512 + (size_t)lane * 16;
        a0 = __builtin_amdgcn_wmma_f32_16x16x32_f16(
            false, ae, false, *(const v16h*)(rf + (size_t)wv * 512), (short)0, a0, false, false);
        a1 = __builtin_amdgcn_wmma_f32_16x16x32_f16(
            false, ae, false, *(const v16h*)(rf + (size_t)(wv + 4) * 512), (short)0, a1, false, false);
    }
    float bi0 = b1[wv * 16 + row];
    float bi1 = b1[(wv + 4) * 16 + row];
    #pragma unroll
    for (int g = 0; g < 8; ++g) {
        int m = g + hi * 8;
        m1f[(size_t)(mt * 16 + m) * D + wv * 16 + row]       = fmaxf(a0[g] + bi0, 0.0f);
        m1f[(size_t)(mt * 16 + m) * D + (wv + 4) * 16 + row] = fmaxf(a1[g] + bi1, 0.0f);
    }
}

// ---------------------------------------------------------------------------
// Kernel 0c: pre-swizzle X = [x_m | m1] (f32) into f16 WMMA B-fragments.
// Layout: xfrag[kk][ct][lane][16 halves], frag = 32x16 (KxN) tile.
// ---------------------------------------------------------------------------
__global__ void k_xfrag(const float* __restrict__ xm, const float* __restrict__ m1f,
                        _Float16* __restrict__ xfrag) {
    int lane = threadIdx.x;               // 0..31
    int b    = blockIdx.x;                // kk*16 + ct
    int kk   = b >> 4;
    int ct   = b & 15;
    int row  = lane & 15;
    int hi   = lane >> 4;
    int col  = ct * 16 + row;             // 0..255
    v16h v;
    #pragma unroll
    for (int h = 0; h < 16; ++h) {
        int k = kk * 32 + kmap(h, hi);    // mole index
        float x = (col < D) ? xm[(size_t)k * D + col]
                            : m1f[(size_t)k * D + (col - D)];
        v[h] = (_Float16)x;
    }
    *(v16h*)(xfrag + ((size_t)b * 32 + lane) * 16) = v;
}

// ---------------------------------------------------------------------------
// Main fused kernel: per 16-entity tile, one streaming pass over the weight
// matrix builds relation masks on the fly and accumulates BOTH layers'
// relation sums via WMMA; epilogue finishes conv1+conv2 entirely on-chip.
// 128 threads = 4 waves; wave w owns column tiles {w, w+4, w+8, w+12} of X,
// so each mask build feeds 20 WMMAs (4x less redundant VALU than 8 waves).
// ---------------------------------------------------------------------------
__global__ void __launch_bounds__(128)
k_main(const float* __restrict__ w,       // [NE, NM] edge weights
       const float* __restrict__ xe,      // [NE, D]
       const _Float16* __restrict__ xfrag,
       const _Float16* __restrict__ wfrag,
       const float* __restrict__ b1,
       const float* __restrict__ b2,
       float* __restrict__ out) {

    __shared__ __align__(16) _Float16 ls1[NR][16][D];   // raw relation sums (conv1 src)
    __shared__ __align__(16) _Float16 ls2[NR][16][D];   // raw relation sums (conv2 src = m1)
    __shared__ __align__(16) _Float16 lse[16][D];       // e1 tile (post-ReLU)
    __shared__ float lcnt[32][NR];                      // per-lane partial counts

    const int tid  = threadIdx.x;
    const int lane = tid & 31;
    const int wv   = tid >> 5;                          // 0..3
    const int row  = lane & 15;
    const int hi   = lane >> 4;
    const int et   = blockIdx.x;                        // entity tile (625)

    // ---------------- Phase A: masked sums over moles ----------------
    v8f acc[4][NR];                                     // ct = wv + 4*t
    #pragma unroll
    for (int t = 0; t < 4; ++t)
        #pragma unroll
        for (int r = 0; r < NR; ++r)
            acc[t][r] = (v8f)(0.0f);

    float cnt[NR] = {0.f, 0.f, 0.f, 0.f, 0.f};

    const float* wrow = w + (size_t)(et * 16 + row) * NM + hi * 8;

    for (int kk = 0; kk < KST; ++kk) {
        const float* p = wrow + kk * 32;
        __builtin_prefetch(p + 4 * 32, 0, 1);           // stream 4 K-steps ahead
        float4 a0 = *(const float4*)(p);
        float4 a1 = *(const float4*)(p + 4);
        float4 c0 = *(const float4*)(p + 16);
        float4 c1 = *(const float4*)(p + 20);
        float vals[16] = {a0.x, a0.y, a0.z, a0.w, a1.x, a1.y, a1.z, a1.w,
                          c0.x, c0.y, c0.z, c0.w, c1.x, c1.y, c1.z, c1.w};
        int li[16];
        #pragma unroll
        for (int h = 0; h < 16; ++h)
            li[h] = (int)ceilf(vals[h] * 6.0f) - 1;     // relation level

        const _Float16* xf = xfrag + (size_t)kk * CT * 512;
        v16h bf[4];
        #pragma unroll
        for (int t = 0; t < 4; ++t)
            bf[t] = *(const v16h*)(xf + ((size_t)(wv + 4 * t) * 32 + lane) * 16);

        #pragma unroll
        for (int r = 0; r < NR; ++r) {                  // one live mask at a time
            v16h mk;
            #pragma unroll
            for (int h = 0; h < 16; ++h)
                mk[h] = (li[h] == r + 1) ? (_Float16)1.0f : (_Float16)0.0f;
            if (wv == 0) {                              // counts once per block
                #pragma unroll
                for (int h = 0; h < 16; ++h)
                    cnt[r] += (li[h] == r + 1) ? 1.0f : 0.0f;
            }
            #pragma unroll
            for (int t = 0; t < 4; ++t)
                acc[t][r] = __builtin_amdgcn_wmma_f32_16x16x32_f16(
                    false, mk, false, bf[t], (short)0, acc[t][r], false, false);
        }
    }

    // stage raw sums + counts to LDS
    #pragma unroll
    for (int t = 0; t < 4; ++t) {
        int ct = wv + 4 * t;
        #pragma unroll
        for (int r = 0; r < NR; ++r) {
            #pragma unroll
            for (int g = 0; g < 8; ++g) {
                int m = g + hi * 8;
                if (ct < 8) ls1[r][m][ct * 16 + row]       = (_Float16)acc[t][r][g];
                else        ls2[r][m][(ct - 8) * 16 + row] = (_Float16)acc[t][r][g];
            }
        }
    }
    if (wv == 0) {
        #pragma unroll
        for (int r = 0; r < NR; ++r) lcnt[lane][r] = cnt[r];
    }
    __syncthreads();

    // per-entity-row 1/deg (masks identical in both conv layers)
    _Float16 invh[NR];
    #pragma unroll
    for (int r = 0; r < NR; ++r) {
        float c = lcnt[row][r] + lcnt[row + 16][r];
        invh[r] = (_Float16)(c > 0.0f ? 1.0f / c : 0.0f);
    }

    // ---------------- Phase B: conv1  e1 = relu(x_e@root1 + Σ agg1_r@W1[r] + b1)
    const float* xerow = xe + (size_t)(et * 16 + row) * D + hi * 8;
    v8f e1[2] = {(v8f)(0.0f), (v8f)(0.0f)};             // output ct = wv, wv+4
    #pragma unroll
    for (int kk = 0; kk < 4; ++kk) {
        v16h ae = load_a_frag_f32(xerow + kk * 32);
        const _Float16* rf = wfrag + (((size_t)10 * 4 + kk) * 8) * 512 + (size_t)lane * 16;
        e1[0] = __builtin_amdgcn_wmma_f32_16x16x32_f16(
            false, ae, false, *(const v16h*)(rf + (size_t)wv * 512), (short)0, e1[0], false, false);
        e1[1] = __builtin_amdgcn_wmma_f32_16x16x32_f16(
            false, ae, false, *(const v16h*)(rf + (size_t)(wv + 4) * 512), (short)0, e1[1], false, false);

        #pragma unroll
        for (int r = 0; r < NR; ++r) {
            v8h lo  = *(const v8h*)&ls1[r][row][kk * 32 + hi * 8];
            v8h hi8 = *(const v8h*)&ls1[r][row][kk * 32 + 16 + hi * 8];
            v16h ar;
            #pragma unroll
            for (int i = 0; i < 8; ++i) {
                ar[i]     = lo[i]  * invh[r];
                ar[i + 8] = hi8[i] * invh[r];
            }
            const _Float16* wf = wfrag + (((size_t)r * 4 + kk) * 8) * 512 + (size_t)lane * 16;
            e1[0] = __builtin_amdgcn_wmma_f32_16x16x32_f16(
                false, ar, false, *(const v16h*)(wf + (size_t)wv * 512), (short)0, e1[0], false, false);
            e1[1] = __builtin_amdgcn_wmma_f32_16x16x32_f16(
                false, ar, false, *(const v16h*)(wf + (size_t)(wv + 4) * 512), (short)0, e1[1], false, false);
        }
    }
    {
        float bi0 = b1[wv * 16 + row];
        float bi1 = b1[(wv + 4) * 16 + row];
        #pragma unroll
        for (int g = 0; g < 8; ++g) {
            int m = g + hi * 8;
            lse[m][wv * 16 + row]       = (_Float16)fmaxf(e1[0][g] + bi0, 0.0f);
            lse[m][(wv + 4) * 16 + row] = (_Float16)fmaxf(e1[1][g] + bi1, 0.0f);
        }
    }
    __syncthreads();

    // ---------------- Phase C: conv2  e2 = e1@root2 + Σ agg2_r@W2[r] + b2
    v8f e2[2] = {(v8f)(0.0f), (v8f)(0.0f)};
    #pragma unroll
    for (int kk = 0; kk < 4; ++kk) {
        v8h lo  = *(const v8h*)&lse[row][kk * 32 + hi * 8];
        v8h hi8 = *(const v8h*)&lse[row][kk * 32 + 16 + hi * 8];
        v16h ae;
        #pragma unroll
        for (int i = 0; i < 8; ++i) { ae[i] = lo[i]; ae[i + 8] = hi8[i]; }

        const _Float16* rf = wfrag + (((size_t)11 * 4 + kk) * 8) * 512 + (size_t)lane * 16;
        e2[0] = __builtin_amdgcn_wmma_f32_16x16x32_f16(
            false, ae, false, *(const v16h*)(rf + (size_t)wv * 512), (short)0, e2[0], false, false);
        e2[1] = __builtin_amdgcn_wmma_f32_16x16x32_f16(
            false, ae, false, *(const v16h*)(rf + (size_t)(wv + 4) * 512), (short)0, e2[1], false, false);

        #pragma unroll
        for (int r = 0; r < NR; ++r) {
            v8h l2 = *(const v8h*)&ls2[r][row][kk * 32 + hi * 8];
            v8h h2 = *(const v8h*)&ls2[r][row][kk * 32 + 16 + hi * 8];
            v16h ar;
            #pragma unroll
            for (int i = 0; i < 8; ++i) {
                ar[i]     = l2[i] * invh[r];
                ar[i + 8] = h2[i] * invh[r];
            }
            const _Float16* wf = wfrag + (((size_t)(5 + r) * 4 + kk) * 8) * 512 + (size_t)lane * 16;
            e2[0] = __builtin_amdgcn_wmma_f32_16x16x32_f16(
                false, ar, false, *(const v16h*)(wf + (size_t)wv * 512), (short)0, e2[0], false, false);
            e2[1] = __builtin_amdgcn_wmma_f32_16x16x32_f16(
                false, ar, false, *(const v16h*)(wf + (size_t)(wv + 4) * 512), (short)0, e2[1], false, false);
        }
    }
    {
        float bi0 = b2[wv * 16 + row];
        float bi1 = b2[(wv + 4) * 16 + row];
        #pragma unroll
        for (int g = 0; g < 8; ++g) {
            int m = g + hi * 8;
            out[(size_t)(et * 16 + m) * D + wv * 16 + row]       = e2[0][g] + bi0;
            out[(size_t)(et * 16 + m) * D + (wv + 4) * 16 + row] = e2[1][g] + bi1;
        }
    }
}

// ---------------------------------------------------------------------------
extern "C" void kernel_launch(void* const* d_in, const int* in_sizes, int n_in,
                              void* d_out, int out_size, void* d_ws, size_t ws_size,
                              hipStream_t stream) {
    const float* emb_entity = (const float*)d_in[0];  // [1, NE, D]
    const float* emb_mole   = (const float*)d_in[1];  // [1, NM, D]
    const float* emw        = (const float*)d_in[2];  // [NE, NM]
    const float* W1         = (const float*)d_in[3];  // [6, D, D]
    const float* root1      = (const float*)d_in[4];  // [D, D]
    const float* b1         = (const float*)d_in[5];  // [D]
    const float* W2         = (const float*)d_in[6];  // [6, D, D]
    const float* root2      = (const float*)d_in[7];  // [D, D]
    const float* b2         = (const float*)d_in[8];  // [D]
    float* out = (float*)d_out;

    char* ws = (char*)d_ws;
    float*     m1f   = (float*)ws;                              // 1 MB
    _Float16*  xfrag = (_Float16*)(ws + (size_t)1048576);       // 1 MB
    _Float16*  wfrag = (_Float16*)(ws + (size_t)2097152);       // 384 KB

    k_wfrag<<<12 * 32, 32, 0, stream>>>(W1, W2, root1, root2, wfrag);
    k_m1   <<<NM / 16, 128, 0, stream>>>(emb_mole, wfrag, b1, m1f);
    k_xfrag<<<KST * CT, 32, 0, stream>>>(emb_mole, m1f, xfrag);
    k_main <<<NE / 16, 128, 0, stream>>>(emw, emb_entity, xfrag, wfrag, b1, b2, out);
}
